// PointNetPlusPlus_AutoEncoder_67345087201886
// MI455X (gfx1250) — compile-verified, matrix operand order checked by reading, not run on hardware
//
#include <hip/hip_runtime.h>
#include <hip/hip_bf16.h>
#include <stddef.h>

typedef __attribute__((ext_vector_type(16))) _Float16 v16h;
typedef __attribute__((ext_vector_type(8)))  float    v8f;

#define ROWS 128     // neighbor-rows staged per workgroup in the fused SA kernel
#define FSTR 356     // feature LDS row stride (halves) >= roundup32(323)=352
#define PSTR 264     // ping LDS row stride (halves)    >= 256

// ---------------------------------------------------------------------------
// WMMA fragment helpers (gfx1250 wave32, V_WMMA_F32_16X16X32_F16).
// A 16x32 (MxK) f16 per ISA 7.12.2:
//   lanes 0-15 : m=lane,    VGPR j in 0..3 -> K=2j,2j+1 ; j in 4..7 -> K=16+..
//   lanes 16-31: m=lane-16, same pattern with K offset +8 / +24.
// B 32x16 (KxN) mirrored (n in place of m): row-major [N][K32] f16 weights
// load with the identical per-lane pattern. All operands are K-padded to a
// multiple of 32 with zeros, so every load below is unconditional (compiler
// merges each 8-element run into a b128 load).
// ---------------------------------------------------------------------------
__device__ __forceinline__ v16h frag_h16(const _Float16* src, int lane,
                                         int row0, int k0, int stride) {
  int m  = row0 + (lane & 15);
  int kb = k0 + ((lane & 16) ? 8 : 0);
  const _Float16* rp = src + (size_t)m * stride;
  v16h a;
#pragma unroll
  for (int j = 0; j < 8; ++j) {
    int k = kb + ((j & 4) << 2) + ((j & 3) << 1);
    a[2*j]   = rp[k];
    a[2*j+1] = rp[k + 1];
  }
  return a;
}

// A fragment from padded row-major f32 (activations in workspace).
__device__ __forceinline__ v16h frag_f32(const float* src, int lane,
                                         int row0, int k0, int stride) {
  int m  = row0 + (lane & 15);
  int kb = k0 + ((lane & 16) ? 8 : 0);
  const float* rp = src + (size_t)m * stride;
  v16h a;
#pragma unroll
  for (int j = 0; j < 8; ++j) {
    int k = kb + ((j & 4) << 2) + ((j & 3) << 1);
    a[2*j]   = (_Float16)rp[k];
    a[2*j+1] = (_Float16)rp[k + 1];
  }
  return a;
}

// One MLP layer inside the fused SA kernel: LDS f16 [ROWS, c_in] -> LDS f16
// [ROWS, c_out], relu(bias+WMMA). 8 waves split N tiles; each wave walks all
// 8 M-tiles, reusing the B fragment across them.
__device__ __forceinline__ void layer_mid(const _Float16* src, _Float16* dst,
                                          const _Float16* Wh, const float* bias,
                                          int c_inL, int c_out, int sstr, int dstr,
                                          int tid) {
  int lane = tid & 31, wave = tid >> 5;
  int K32 = (c_inL + 31) & ~31;
  int NT  = c_out >> 4;
  for (int nt = wave; nt < NT; nt += 8) {
    float bn = bias[nt * 16 + (lane & 15)];
    v8f acc[8] = {};
    for (int k0 = 0; k0 < K32; k0 += 32) {
      v16h bf = frag_h16(Wh, lane, nt * 16, k0, K32);
#pragma unroll
      for (int mt = 0; mt < 8; ++mt) {
        v16h af = frag_h16(src, lane, mt * 16, k0, sstr);
        acc[mt] = __builtin_amdgcn_wmma_f32_16x16x32_f16(
            false, af, false, bf, (short)0, acc[mt], false, false);
      }
    }
    int col = nt * 16 + (lane & 15);
    int rof = (lane & 16) ? 8 : 0;
#pragma unroll
    for (int mt = 0; mt < 8; ++mt)
#pragma unroll
      for (int v = 0; v < 8; ++v) {
        int row = mt * 16 + rof + v;
        dst[row * dstr + col] = (_Float16)fmaxf(acc[mt][v] + bn, 0.f);
      }
  }
}

// Final layer + max-pool over the K_nb neighbors of each centroid, done in
// registers: in-lane max over 8 D VGPRs (8 rows) + shfl_xor(16) across wave
// halves; a centroid group is (K_nb/16) consecutive 16-row tiles.
__device__ __forceinline__ void layer_final(const _Float16* src, float* outp,
                                            const _Float16* Wh, const float* bias,
                                            int c_inL, int c_out, int sstr, int tid,
                                            int Knb, int b, int S, int s0,
                                            int c_total, int c_off) {
  int lane = tid & 31, wave = tid >> 5;
  int K32 = (c_inL + 31) & ~31;
  int NT  = c_out >> 4;
  int tpc = Knb >> 4;                         // 16-row tiles per centroid (1,2,4,8)
  for (int nt = wave; nt < NT; nt += 8) {
    float bn = bias[nt * 16 + (lane & 15)];
    v8f acc[8] = {};
    for (int k0 = 0; k0 < K32; k0 += 32) {
      v16h bf = frag_h16(Wh, lane, nt * 16, k0, K32);
#pragma unroll
      for (int mt = 0; mt < 8; ++mt) {
        v16h af = frag_h16(src, lane, mt * 16, k0, sstr);
        acc[mt] = __builtin_amdgcn_wmma_f32_16x16x32_f16(
            false, af, false, bf, (short)0, acc[mt], false, false);
      }
    }
    float gmax = -3.0e38f;
#pragma unroll
    for (int mt = 0; mt < 8; ++mt) {
      float t = acc[mt][0];
#pragma unroll
      for (int v = 1; v < 8; ++v) t = fmaxf(t, acc[mt][v]);
      t = fmaxf(t, __shfl_xor(t, 16, 32));    // combine rows 0-7 with 8-15
      gmax = fmaxf(gmax, t);
      if (((mt + 1) & (tpc - 1)) == 0) {      // centroid group complete
        if ((lane & 16) == 0) {
          int sg = mt / tpc;
          outp[((size_t)(b * S + s0 + sg)) * c_total + c_off + nt * 16 + (lane & 15)] =
              fmaxf(gmax + bn, 0.f);
        }
        gmax = -3.0e38f;
      }
    }
  }
}

// ---------------------------------------------------------------------------
// Fused SA branch: gather(+center) -> MLP x3 (WMMA) -> max over K -> out slice
// ---------------------------------------------------------------------------
__global__ __launch_bounds__(256) void sa_branch_kernel(
    const float* __restrict__ xyz, const float* __restrict__ pts,
    const float* __restrict__ nxyz, const int* __restrict__ gidx,
    const _Float16* __restrict__ W1, const float* __restrict__ b1,
    const _Float16* __restrict__ W2, const float* __restrict__ b2,
    const _Float16* __restrict__ W3, const float* __restrict__ b3,
    float* __restrict__ outp,
    int N, int S, int Knb, int c_pts, int c1, int c2, int c3,
    int c_total, int c_off) {
  extern __shared__ char smem[];
  _Float16* featb = (_Float16*)smem;                                    // [ROWS][FSTR]
  _Float16* pingb = (_Float16*)(smem + (size_t)ROWS * FSTR * 2);        // [ROWS][PSTR]
  int*      gsh   = (int*)(smem + (size_t)ROWS * FSTR * 2 + (size_t)ROWS * PSTR * 2);
  float*    csh   = (float*)(gsh + ROWS);                               // [<=8][3]

  int tid = threadIdx.x;
  int b   = blockIdx.y;
  int TS  = ROWS / Knb;
  int s0  = blockIdx.x * TS;
  int c_in  = c_pts + 3;
  int K32in = (c_in + 31) & ~31;

  if (tid < ROWS)
    gsh[tid] = gidx[((size_t)(b * S + s0 + tid / Knb)) * Knb + (tid % Knb)];
  if (tid < TS * 3)
    csh[tid] = nxyz[((size_t)(b * S + s0)) * 3 + tid];
  __syncthreads();

  // gather features into LDS (points feats first, then centered xyz), zero-pad K
  for (int e = tid; e < ROWS * K32in; e += 256) {
    int r = e / K32in, c = e - r * K32in;
    int p = gsh[r];
    float v = 0.f;
    if (c < c_pts)     v = pts[((size_t)b * N + p) * c_pts + c];
    else if (c < c_in) v = xyz[((size_t)b * N + p) * 3 + (c - c_pts)] -
                           csh[(r / Knb) * 3 + (c - c_pts)];
    featb[r * FSTR + c] = (_Float16)v;
  }
  __syncthreads();

  layer_mid(featb, pingb, W1, b1, c_in, c1, FSTR, PSTR, tid);
  __syncthreads();
  layer_mid(pingb, featb, W2, b2, c1, c2, PSTR, FSTR, tid);   // reuse featb as pong
  __syncthreads();
  layer_final(featb, outp, W3, b3, c2, c3, FSTR, tid, Knb, b, S, s0, c_total, c_off);
}

// ---------------------------------------------------------------------------
// Generic C = act(A[M,K32] @ Wh[N,K32]^T + bias), one wave per 16x16 tile.
// A: f32 with padded stride lda (rows padded to 16, pad cols zeroed by the
// producer); Wh: padded f16. All fragment loads unconditional.
// ---------------------------------------------------------------------------
__global__ void wmma_gemm_kernel(const float* __restrict__ A, int lda,
                                 const _Float16* __restrict__ Wh,
                                 const float* __restrict__ bias, float* __restrict__ C,
                                 int M, int N, int K32, int relu) {
  int lane = threadIdx.x & 31;
  int m0 = blockIdx.x * 16, n0 = blockIdx.y * 16;
  v8f acc = {};
  for (int k0 = 0; k0 < K32; k0 += 32) {
    v16h af = frag_f32(A, lane, m0, k0, lda);
    v16h bf = frag_h16(Wh, lane, n0, k0, K32);
    acc = __builtin_amdgcn_wmma_f32_16x16x32_f16(false, af, false, bf, (short)0, acc,
                                                 false, false);
  }
  int n = n0 + (lane & 15);
  float bn = bias[n];
  int rof = (lane & 16) ? 8 : 0;
#pragma unroll
  for (int v = 0; v < 8; ++v) {
    int m = m0 + rof + v;
    if (m < M) {
      float tv = acc[v] + bn;
      if (relu) tv = fmaxf(tv, 0.f);
      C[(size_t)m * N + n] = tv;
    }
  }
}

// Weight prep: f32 [N][K] -> f16 [N][K32], zero padded.
__global__ void wprep_kernel(const float* __restrict__ W, _Float16* __restrict__ Wh,
                             int K, int K32, int total) {
  int t = blockIdx.x * blockDim.x + threadIdx.x;
  if (t >= total) return;
  int n = t / K32, k = t - n * K32;
  Wh[t] = (k < K) ? (_Float16)W[(size_t)n * K + k] : (_Float16)0.f;
}

// ---------------------------------------------------------------------------
// Farthest point sampling: one workgroup per batch; LDS argmax tree with
// first-index tie-break (matches jnp.argmax). blockDim.x == N.
// ---------------------------------------------------------------------------
__global__ __launch_bounds__(1024) void fps_kernel(const float* __restrict__ xyz,
                                                   int* __restrict__ fidx, int N, int S) {
  __shared__ float sval[1024];
  __shared__ int   sidx[1024];
  __shared__ float cur[3];
  int b = blockIdx.x, tid = threadIdx.x;
  const float* base = xyz + (size_t)b * N * 3;
  float px = base[tid * 3 + 0], py = base[tid * 3 + 1], pz = base[tid * 3 + 2];
  float dist = 1e10f;
  int far = 0;
  for (int it = 0; it < S; ++it) {
    if (tid == 0) {
      fidx[(size_t)b * S + it] = far;
      cur[0] = base[far * 3 + 0]; cur[1] = base[far * 3 + 1]; cur[2] = base[far * 3 + 2];
    }
    __syncthreads();
    float dx = px - cur[0], dy = py - cur[1], dz = pz - cur[2];
    dist = fminf(dist, dx * dx + dy * dy + dz * dz);
    sval[tid] = dist; sidx[tid] = tid;
    __syncthreads();
    for (int off = N >> 1; off > 0; off >>= 1) {
      if (tid < off) {
        float v2 = sval[tid + off]; int i2 = sidx[tid + off];
        if (v2 > sval[tid] || (v2 == sval[tid] && i2 < sidx[tid])) {
          sval[tid] = v2; sidx[tid] = i2;
        }
      }
      __syncthreads();
    }
    far = sidx[0];
  }
}

__global__ void gather_xyz_kernel(const float* __restrict__ xyz, const int* __restrict__ fidx,
                                  float* __restrict__ nxyz, int N, int S, int total) {
  int t = blockIdx.x * blockDim.x + threadIdx.x;
  if (t >= total) return;
  int b = t / S;
  int p = fidx[t];
  const float* src = xyz + ((size_t)b * N + p) * 3;
  float* dst = nxyz + (size_t)t * 3;
  dst[0] = src[0]; dst[1] = src[1]; dst[2] = src[2];
}

// yanx27 query_ball_point: first K in-radius indices in ascending order,
// padded with the first hit.
__global__ void ballq_kernel(const float* __restrict__ xyz, const float* __restrict__ nxyz,
                             int* __restrict__ gidx, int N, int S, int K, float r2,
                             int total) {
  int t = blockIdx.x * blockDim.x + threadIdx.x;
  if (t >= total) return;
  int b = t / S;
  const float* base = xyz + (size_t)b * N * 3;
  float cx = nxyz[(size_t)t * 3 + 0], cy = nxyz[(size_t)t * 3 + 1], cz = nxyz[(size_t)t * 3 + 2];
  int* out = gidx + (size_t)t * K;
  int cnt = 0;
  for (int n = 0; n < N && cnt < K; ++n) {
    float dx = base[n * 3 + 0] - cx, dy = base[n * 3 + 1] - cy, dz = base[n * 3 + 2] - cz;
    if (dx * dx + dy * dy + dz * dz <= r2) out[cnt++] = n;
  }
  int first = (cnt > 0) ? out[0] : 0;
  for (int j = cnt; j < K; ++j) out[j] = first;
}

// concat [l2_xyz, l2_pts] into padded stride-672 buffer (zeros in pad cols).
__global__ void concat3_kernel(const float* __restrict__ nx2, const float* __restrict__ l2p,
                               float* __restrict__ f3, int total) {
  int t = blockIdx.x * blockDim.x + threadIdx.x;
  if (t >= total) return;
  int row = t / 672, c = t - row * 672;
  float v = 0.f;
  if (c < 3)        v = nx2[(size_t)row * 3 + c];
  else if (c < 643) v = l2p[(size_t)row * 640 + (c - 3)];
  f3[t] = v;
}

__global__ void gmax_kernel(const float* __restrict__ h, float* __restrict__ g,
                            int rowsper, int C, int total) {
  int t = blockIdx.x * blockDim.x + threadIdx.x;
  if (t >= total) return;
  int b = t / C, c = t - b * C;
  const float* p = h + (size_t)b * rowsper * C + c;
  float m = p[0];
  for (int i = 1; i < rowsper; ++i) m = fmaxf(m, p[(size_t)i * C]);
  g[t] = m;
}

// h[B,3072] viewed as (B,3,1024) -> out (B,1024,3)
__global__ void out_transpose_kernel(const float* __restrict__ h, float* __restrict__ out,
                                     int total) {
  int t = blockIdx.x * blockDim.x + threadIdx.x;
  if (t >= total) return;
  int b = t / 3072, r = t - b * 3072;
  int n = r / 3, c = r - n * 3;
  out[t] = h[(size_t)b * 3072 + c * 1024 + n];
}

// ---------------------------------------------------------------------------
extern "C" void kernel_launch(void* const* d_in, const int* in_sizes, int n_in,
                              void* d_out, int out_size, void* d_ws, size_t ws_size,
                              hipStream_t stream) {
  (void)n_in; (void)out_size; (void)ws_size;

  // Input mapping: insertion order (x, sa1, sa2, sa3, lin1, lin2, dec) or
  // sorted-pytree order (dec, lin1, lin2, sa1, sa2, sa3, x) — detect by leaf 0.
  bool alpha = (in_sizes[0] != 12288);
  int i_x, i_sa1, i_sa2, i_sa3, i_lin1, i_lin2, i_dec;
  if (!alpha) { i_x = 0; i_sa1 = 1;  i_sa2 = 19; i_sa3 = 37; i_lin1 = 43; i_lin2 = 45; i_dec = 47; }
  else        { i_dec = 0; i_lin1 = 6; i_lin2 = 8; i_sa1 = 10; i_sa2 = 28; i_sa3 = 46; i_x = 52; }
  auto F = [&](int i) { return (const float*)d_in[i]; };
  const float* x = F(i_x);
  int B = in_sizes[i_x] / (1024 * 3);

  // workspace carve
  char* wsp = (char*)d_ws;
  size_t off = 0;
  auto alloc = [&](size_t bytes) {
    void* p = wsp + off;
    off = (off + bytes + 255) & ~(size_t)255;
    return p;
  };
  int*   fidx1 = (int*)  alloc((size_t)B * 512 * 4);
  float* nx1   = (float*)alloc((size_t)B * 512 * 3 * 4);
  int*   gidx  = (int*)  alloc((size_t)B * 512 * 128 * 4);
  float* l1p   = (float*)alloc((size_t)B * 512 * 320 * 4);
  int*   fidx2 = (int*)  alloc((size_t)B * 128 * 4);
  float* nx2   = (float*)alloc((size_t)B * 128 * 3 * 4);
  float* l2p   = (float*)alloc((size_t)B * 128 * 640 * 4);
  float* f3    = (float*)alloc((size_t)B * 128 * 672 * 4);   // padded K stride
  float* h1    = (float*)alloc((size_t)B * 128 * 256 * 4);
  float* h2    = (float*)alloc((size_t)B * 128 * 512 * 4);
  float* h3    = (float*)alloc((size_t)B * 128 * 1024 * 4);
  float* g     = (float*)alloc((size_t)16 * 1024 * 4);       // rows padded to 16
  float* e1    = (float*)alloc((size_t)16 * 512 * 4);
  float* emb   = (float*)alloc((size_t)16 * 512 * 4);
  float* dd1   = (float*)alloc((size_t)16 * 512 * 4);
  float* dd2   = (float*)alloc((size_t)16 * 1024 * 4);
  float* dd3   = (float*)alloc((size_t)16 * 3072 * 4);

  // f16 K-padded weight prep (runs every call; in-order on stream).
  auto prep = [&](int i, int N, int K) -> const _Float16* {
    int K32 = (K + 31) & ~31;
    _Float16* p = (_Float16*)alloc((size_t)N * K32 * 2);
    int tot = N * K32;
    wprep_kernel<<<(tot + 255) / 256, 256, 0, stream>>>(F(i), p, K, K32, tot);
    return p;
  };

  size_t smem = (size_t)ROWS * FSTR * 2 + (size_t)ROWS * PSTR * 2 + ROWS * 4 + 8 * 3 * 4;

  auto gemm = [&](const float* A, int lda, const _Float16* Wh, const float* bias,
                  float* Cm, int M, int N, int K32, int relu) {
    dim3 grid((M + 15) / 16, N / 16);
    wmma_gemm_kernel<<<grid, dim3(32), 0, stream>>>(A, lda, Wh, bias, Cm, M, N, K32, relu);
  };

  const int   K1[3]    = {16, 32, 128};
  const float R1[3]    = {0.1f, 0.2f, 0.4f};
  const int   C1[3][3] = {{32, 32, 64}, {64, 64, 128}, {64, 96, 128}};
  const int   O1[3]    = {0, 64, 192};
  const int   K2[3]    = {32, 64, 128};
  const float R2[3]    = {0.2f, 0.4f, 0.8f};
  const int   C2[3][3] = {{64, 64, 128}, {128, 128, 256}, {128, 128, 256}};
  const int   O2[3]    = {0, 128, 384};

  // ---- SA1: FPS 1024->512, 3 MSG branches, out l1p [B,512,320]
  fps_kernel<<<B, 1024, 0, stream>>>(x, fidx1, 1024, 512);
  { int tot = B * 512;
    gather_xyz_kernel<<<(tot + 255) / 256, 256, 0, stream>>>(x, fidx1, nx1, 1024, 512, tot); }
  for (int br = 0; br < 3; ++br) {
    int Kn = K1[br], wi = i_sa1 + br * 6;
    const _Float16* w1 = prep(wi,     C1[br][0], 3);
    const _Float16* w2 = prep(wi + 2, C1[br][1], C1[br][0]);
    const _Float16* w3 = prep(wi + 4, C1[br][2], C1[br][1]);
    { int tot = B * 512;
      ballq_kernel<<<(tot + 255) / 256, 256, 0, stream>>>(x, nx1, gidx, 1024, 512, Kn,
                                                          R1[br] * R1[br], tot); }
    dim3 grid(512 * Kn / ROWS, B);
    sa_branch_kernel<<<grid, 256, smem, stream>>>(
        x, nullptr, nx1, gidx, w1, F(wi + 1), w2, F(wi + 3), w3, F(wi + 5),
        l1p, 1024, 512, Kn, 0, C1[br][0], C1[br][1], C1[br][2], 320, O1[br]);
  }

  // ---- SA2: FPS 512->128, 3 MSG branches (c_in = 320+3), out l2p [B,128,640]
  fps_kernel<<<B, 512, 0, stream>>>(nx1, fidx2, 512, 128);
  { int tot = B * 128;
    gather_xyz_kernel<<<(tot + 255) / 256, 256, 0, stream>>>(nx1, fidx2, nx2, 512, 128, tot); }
  for (int br = 0; br < 3; ++br) {
    int Kn = K2[br], wi = i_sa2 + br * 6;
    const _Float16* w1 = prep(wi,     C2[br][0], 323);
    const _Float16* w2 = prep(wi + 2, C2[br][1], C2[br][0]);
    const _Float16* w3 = prep(wi + 4, C2[br][2], C2[br][1]);
    { int tot = B * 128;
      ballq_kernel<<<(tot + 255) / 256, 256, 0, stream>>>(nx1, nx2, gidx, 512, 128, Kn,
                                                          R2[br] * R2[br], tot); }
    dim3 grid(128 * Kn / ROWS, B);
    sa_branch_kernel<<<grid, 256, smem, stream>>>(
        nx1, l1p, nx2, gidx, w1, F(wi + 1), w2, F(wi + 3), w3, F(wi + 5),
        l2p, 512, 128, Kn, 320, C2[br][0], C2[br][1], C2[br][2], 640, O2[br]);
  }

  // ---- SA3 (group_all): concat xyz+feats, MLP 643->256->512->1024, global max
  const _Float16* ws3a = prep(i_sa3,     256, 643);
  const _Float16* ws3b = prep(i_sa3 + 2, 512, 256);
  const _Float16* ws3c = prep(i_sa3 + 4, 1024, 512);
  const _Float16* wl1  = prep(i_lin1,    512, 1024);
  const _Float16* wl2  = prep(i_lin2,    512, 512);
  const _Float16* wd1  = prep(i_dec,     512, 512);
  const _Float16* wd2  = prep(i_dec + 2, 1024, 512);
  const _Float16* wd3  = prep(i_dec + 4, 3072, 1024);

  { int tot = B * 128 * 672;
    concat3_kernel<<<(tot + 255) / 256, 256, 0, stream>>>(nx2, l2p, f3, tot); }
  gemm(f3, 672,  ws3a, F(i_sa3 + 1), h1, B * 128, 256, 672, 1);
  gemm(h1, 256,  ws3b, F(i_sa3 + 3), h2, B * 128, 512, 256, 1);
  gemm(h2, 512,  ws3c, F(i_sa3 + 5), h3, B * 128, 1024, 512, 1);
  { int tot = B * 1024;
    gmax_kernel<<<(tot + 255) / 256, 256, 0, stream>>>(h3, g, 128, 1024, tot); }

  // ---- FC encoder/decoder (M=B rows, padded allocations to 16 rows)
  gemm(g,   1024, wl1, F(i_lin1 + 1), e1,  B, 512, 1024, 1);
  gemm(e1,  512,  wl2, F(i_lin2 + 1), emb, B, 512, 512, 0);
  gemm(emb, 512,  wd1, F(i_dec + 1),  dd1, B, 512, 512, 1);
  gemm(dd1, 512,  wd2, F(i_dec + 3),  dd2, B, 1024, 512, 1);
  gemm(dd2, 1024, wd3, F(i_dec + 5),  dd3, B, 3072, 1024, 0);

  { int tot = B * 3072;
    out_transpose_kernel<<<(tot + 255) / 256, 256, 0, stream>>>(dd3, (float*)d_out, tot); }
}
